// LeWinTransformerBlock_17927193493651
// MI455X (gfx1250) — matrix operand order repaired and checked by reading, not compiled, not run on hardware
//
#include <hip/hip_runtime.h>
#include <hip/hip_bf16.h>

typedef __attribute__((ext_vector_type(16))) _Float16 v16h;
typedef __attribute__((ext_vector_type(8)))  _Float16 h8;
typedef __attribute__((ext_vector_type(4)))  _Float16 h4;
typedef __attribute__((ext_vector_type(8)))  float    v8f;

#define NWIN_TOTAL 4096      // B * (H/8) * (W/8)
#define CC 64                // channels
#define NHEADS 4
#define HDIM 16

// ---- dynamic LDS layout (bytes) ----
#define OFF_XW 0             // float  [64][64]   residual / x2        16384
#define OFF_HW 16384         // f16    [64][72]   LN out / attn out     9216
#define OFF_Q  25600         // f16    [64][72]   q (row major)         9216
#define OFF_K  34816         // f16    [64][72]   k (row major)         9216
#define OFF_V  44032         // f16    [64][72]   v^T (chan major)      9216
#define OFF_S  53248         // float  [4][64][68] bias/scores/probs/gelu 69632
#define OFF_WT 122880        // f16 transposed weight stage (<=20480 halves) 40960
#define SMEM_BYTES 163840

static __device__ __forceinline__ v8f wmma32(v16h a, v16h b, v8f c) {
  return __builtin_amdgcn_wmma_f32_16x16x32_f16(false, a, false, b, (short)0, c,
                                                false, false);
}

static __device__ __forceinline__ v16h cat16(h8 lo, h8 hi) {
  return __builtin_shufflevector(lo, hi, 0, 1, 2, 3, 4, 5, 6, 7,
                                 8, 9, 10, 11, 12, 13, 14, 15);
}

// Async DMA copy: 16 bytes global -> LDS per lane (gfx1250 async path).
static __device__ __forceinline__ void async_copy16(void* lds_dst,
                                                    const void* gsrc) {
  unsigned lds_off = (unsigned)(unsigned long long)lds_dst;
  unsigned long long ga = (unsigned long long)gsrc;
  asm volatile("global_load_async_to_lds_b128 %0, %1, off"
               :: "v"(lds_off), "v"(ga)
               : "memory");
}

static __device__ __forceinline__ void wait_async0() {
  asm volatile("s_wait_asynccnt 0x0" ::: "memory");
}

// A fragment: rows r0..r0+15, K = k0..k0+31, row-major f16 [rows][stride].
static __device__ __forceinline__ v16h load_a16(const _Float16* base, int r0,
                                                int k0, int stride, int lane) {
  int m = lane & 15;
  int kb = (lane & 16) ? 8 : 0;
  const _Float16* p = base + (r0 + m) * stride + k0 + kb;
  h8 lo = *(const h8*)p;          // K = kb+0..7
  h8 hi = *(const h8*)(p + 16);   // K = 16+kb+0..7
  return cat16(lo, hi);
}

// A fragment, true K=16 zero-padded to 32
static __device__ __forceinline__ v16h load_a16_k16(const _Float16* base, int r0,
                                                    int k0, int stride, int lane) {
  int m = lane & 15;
  int kb = (lane & 16) ? 8 : 0;
  const _Float16* p = base + (r0 + m) * stride + k0 + kb;
  h8 lo = *(const h8*)p;
  h8 z = {};
  return cat16(lo, z);
}

// B fragment from K-contiguous (transposed) storage baseT[col][k]:
// element (K,N) = baseT[(c0+N)*strideT + k0 + K]
static __device__ __forceinline__ v16h load_bT(const _Float16* baseT, int c0,
                                               int k0, int strideT, int lane) {
  int n = lane & 15;
  int ko = (lane & 16) ? 8 : 0;   // lanes 16-31 hold K=8..15, 24..31
  const _Float16* p = baseT + (c0 + n) * strideT + k0 + ko;
  h8 lo = *(const h8*)p;
  h8 hi = *(const h8*)(p + 16);
  return cat16(lo, hi);
}

// B fragment for the K=16 score contraction. A is zero for K>=16, so the
// K=16..31 values (elements 8..15 of every lane) are don't-cares (must be finite,
// guaranteed by zeroing the k-buffer pad columns). No divergent branch.
static __device__ __forceinline__ v16h load_bT_k16(const _Float16* baseT, int c0,
                                                   int k0, int strideT, int lane) {
  return load_bT(baseT, c0, k0, strideT, lane);
}

// C fragment (f32) from row-major [rows][stride]
static __device__ __forceinline__ v8f load_c32(const float* base, int r0, int c0,
                                               int stride, int lane) {
  int n = lane & 15;
  int mb = (lane & 16) ? 8 : 0;
  v8f c;
#pragma unroll
  for (int r = 0; r < 8; ++r) c[r] = base[(r0 + mb + r) * stride + c0 + n];
  return c;
}

__global__ void __launch_bounds__(256)
lewin_block_kernel(const float* __restrict__ x,
                   const float* __restrict__ ln1_g, const float* __restrict__ ln1_b,
                   const float* __restrict__ wq,    const float* __restrict__ bq,
                   const float* __restrict__ wkv,   const float* __restrict__ bkv,
                   const float* __restrict__ rbt,   // (225, 4)
                   const float* __restrict__ wo,    const float* __restrict__ bo,
                   const float* __restrict__ ln2_g, const float* __restrict__ ln2_b,
                   const float* __restrict__ w1,    const float* __restrict__ b1,
                   const float* __restrict__ w2,    const float* __restrict__ b2,
                   const int*   __restrict__ rel_index,
                   float* __restrict__ out) {
  extern __shared__ char smem[];
  float*    xw = (float*)(smem + OFF_XW);
  _Float16* hw = (_Float16*)(smem + OFF_HW);
  _Float16* qb = (_Float16*)(smem + OFF_Q);
  _Float16* kbuf = (_Float16*)(smem + OFF_K);
  _Float16* vT = (_Float16*)(smem + OFF_V);
  float*    sb = (float*)(smem + OFF_S);
  _Float16* wt = (_Float16*)(smem + OFF_WT);

  const int tid  = threadIdx.x;
  const int lane = tid & 31;
  // Force wave id into an SGPR so tile loops & epilogue branches are scalar.
  const int wave = __builtin_amdgcn_readfirstlane(tid >> 5);
  const int n    = lane & 15;
  const int mb   = (lane & 16) ? 8 : 0;

  const int wid   = blockIdx.x;
  const int batch = wid >> 10;
  const int wy    = (wid >> 5) & 31;
  const int wx    = wid & 31;

  // warm L2/WGP$ for the MLP weights used in later phases
#pragma unroll
  for (int pf = 0; pf < 4; ++pf) {
    __builtin_prefetch(w1 + ((size_t)tid + pf * 256) * 16, 0, 1);
    __builtin_prefetch(w2 + ((size_t)tid + pf * 256) * 16, 0, 1);
  }

  // ---- Phase A: async-DMA the window into LDS; stage weights/bias; LN1 ----
  {
    int tok = tid >> 2, qtr = tid & 3;
    int ty = tok >> 3, tx = tok & 7;
    size_t gbase = (((size_t)batch << 16) +
                    (size_t)((wy * 8 + ty) * 256 + (wx * 8 + tx))) * CC + qtr * 16;

    // 1) kick off the global->LDS async copies (4 x 16B per thread = 16KB/block)
#pragma unroll
    for (int v4 = 0; v4 < 4; ++v4)
      async_copy16(xw + tok * CC + qtr * 16 + v4 * 4, x + gbase + v4 * 4);

    // 2) overlap: stage fused [wq | wkv] transposed: wt[col][k], stride 72
    for (int i = tid; i < 64 * 192; i += 256) {
      int r = i / 192, c = i % 192;                 // r = K, c = out col
      float v = (c < 64) ? wq[r * 64 + c] : wkv[r * 128 + (c - 64)];
      wt[c * 72 + r] = (_Float16)v;
    }
    // 3) overlap: pre-gather relative-position bias into score buffer (C operand)
    for (int i = tid; i < 64 * 64; i += 256) {
      int ridx = rel_index[i];
      int ii = i >> 6, jj = i & 63;
#pragma unroll
      for (int h = 0; h < NHEADS; ++h)
        sb[h * 64 * 68 + ii * 68 + jj] = rbt[ridx * NHEADS + h];
    }
    // 4) zero k-buffer pad columns (finite don't-care data for score B loads)
    if (tid < 64) {
      h8 z = {};
      *(h8*)(kbuf + tid * 72 + 64) = z;
    }

    // 5) wait for the DMA, then LN1 straight out of LDS
    wait_async0();
    float vals[16];
    float s = 0.f, ss = 0.f;
    const float4* xr = (const float4*)(xw + tok * CC + qtr * 16);
#pragma unroll
    for (int v4 = 0; v4 < 4; ++v4) {
      float4 f = xr[v4];
      vals[v4 * 4 + 0] = f.x; vals[v4 * 4 + 1] = f.y;
      vals[v4 * 4 + 2] = f.z; vals[v4 * 4 + 3] = f.w;
    }
#pragma unroll
    for (int i = 0; i < 16; ++i) { s += vals[i]; ss += vals[i] * vals[i]; }
    s  += __shfl_xor(s, 1);  s  += __shfl_xor(s, 2);
    ss += __shfl_xor(ss, 1); ss += __shfl_xor(ss, 2);
    float mean = s * (1.f / 64.f);
    float var  = ss * (1.f / 64.f) - mean * mean;
    float rstd = rsqrtf(var + 1e-5f);
    h8 o0, o1;
#pragma unroll
    for (int i = 0; i < 8; ++i) {
      int c = qtr * 16 + i;
      o0[i] = (_Float16)((vals[i] - mean) * rstd * ln1_g[c] + ln1_b[c]);
    }
#pragma unroll
    for (int i = 0; i < 8; ++i) {
      int c = qtr * 16 + 8 + i;
      o1[i] = (_Float16)((vals[8 + i] - mean) * rstd * ln1_g[c] + ln1_b[c]);
    }
    *(h8*)(hw + tok * 72 + qtr * 16) = o0;
    *(h8*)(hw + tok * 72 + qtr * 16 + 8) = o1;
  }
  __syncthreads();

  // ---- Phase B: QKV projection (64x64 @ 64x192), scalar tile epilogue ----
  for (int t = wave; t < 48; t += 8) {
    int rt = t / 12, ct = t % 12;          // scalar (t is SGPR)
    int r0 = rt * 16, c0 = ct * 16;
    v8f acc = {};
#pragma unroll
    for (int k0 = 0; k0 < 64; k0 += 32) {
      v16h a = load_a16(hw, r0, k0, 72, lane);
      v16h b = load_bT(wt, c0, k0, 72, lane);
      acc = wmma32(a, b, acc);
    }
    if (ct < 4) {                       // q, scaled by hd^-0.5
      float bias = bq[c0 + n];
#pragma unroll
      for (int r = 0; r < 8; ++r)
        qb[(r0 + mb + r) * 72 + c0 + n] = (_Float16)((acc[r] + bias) * 0.25f);
    } else if (ct < 8) {                // k
      int dc = c0 - 64 + n;
      float bias = bkv[dc];
#pragma unroll
      for (int r = 0; r < 8; ++r)
        kbuf[(r0 + mb + r) * 72 + dc] = (_Float16)(acc[r] + bias);
    } else {                            // v, stored transposed [chan][token]
      int dc = c0 - 128 + n;
      float bias = bkv[64 + dc];
      h8 vv;
#pragma unroll
      for (int r = 0; r < 8; ++r) vv[r] = (_Float16)(acc[r] + bias);
      *(h8*)(vT + dc * 72 + r0 + mb) = vv;
    }
  }
  __syncthreads();

  // ---- Phase C: S = q*scale @ k^T + rel_bias (bias as WMMA C); stage wo^T ----
  for (int i = tid; i < 64 * 64; i += 256) {
    int r = i >> 6, c = i & 63;                     // r = K, c = out col
    wt[c * 72 + r] = (_Float16)wo[i];
  }
  for (int t = wave; t < 64; t += 8) {
    int head = t >> 4, rem = t & 15;
    int r0 = (rem >> 2) * 16, c0 = (rem & 3) * 16;
    float* sh = sb + head * 64 * 68;
    v16h a = load_a16_k16(qb, r0, head * HDIM, 72, lane);
    v16h b = load_bT_k16(kbuf, c0, head * HDIM, 72, lane);
    v8f acc = load_c32(sh, r0, c0, 68, lane);       // rel-pos bias
    acc = wmma32(a, b, acc);
#pragma unroll
    for (int r = 0; r < 8; ++r)
      sh[(r0 + mb + r) * 68 + c0 + n] = acc[r];
  }
  __syncthreads();

  // ---- softmax: 256 threads == 4 heads x 64 rows, vectorized, in-place ----
  {
    int head = tid >> 6, row = tid & 63;
    float* rp = sb + head * 64 * 68 + row * 68;
    float4* rp4 = (float4*)rp;
    float mx = -1e30f;
#pragma unroll
    for (int j = 0; j < 16; ++j) {
      float4 f = rp4[j];
      mx = fmaxf(mx, fmaxf(fmaxf(f.x, f.y), fmaxf(f.z, f.w)));
    }
    float sum = 0.f;
#pragma unroll
    for (int j = 0; j < 16; ++j) {
      float4 f = rp4[j];
      f.x = __expf(f.x - mx); f.y = __expf(f.y - mx);
      f.z = __expf(f.z - mx); f.w = __expf(f.w - mx);
      rp4[j] = f;
      sum += f.x + f.y + f.z + f.w;
    }
    float inv = 1.f / sum;
    h4* hp = (h4*)rp;                 // in-place f32 -> f16 (sequential-safe)
#pragma unroll
    for (int j = 0; j < 16; ++j) {
      float4 f = rp4[j];
      h4 o;
      o[0] = (_Float16)(f.x * inv); o[1] = (_Float16)(f.y * inv);
      o[2] = (_Float16)(f.z * inv); o[3] = (_Float16)(f.w * inv);
      hp[j] = o;
    }
  }
  __syncthreads();

  // ---- Phase D: O = P @ V (per-head 64x64 @ 64x16) ----
  for (int t = wave; t < 16; t += 8) {
    int head = t >> 2, rt = t & 3;
    int r0 = rt * 16;
    const _Float16* pbase = (const _Float16*)(sb + head * 64 * 68); // stride 136
    v8f acc = {};
#pragma unroll
    for (int k0 = 0; k0 < 64; k0 += 32) {
      v16h a = load_a16(pbase, r0, k0, 136, lane);
      v16h b = load_bT(vT, head * HDIM, k0, 72, lane);
      acc = wmma32(a, b, acc);
    }
#pragma unroll
    for (int r = 0; r < 8; ++r)
      hw[(r0 + mb + r) * 72 + head * HDIM + n] = (_Float16)acc[r];
  }
  __syncthreads();

  // ---- Phase E: o-proj + residual into xw ----
  for (int t = wave; t < 16; t += 8) {
    int rt = t >> 2, ct = t & 3;
    int r0 = rt * 16, c0 = ct * 16;
    v8f acc = {};
#pragma unroll
    for (int k0 = 0; k0 < 64; k0 += 32) {
      v16h a = load_a16(hw, r0, k0, 72, lane);
      v16h b = load_bT(wt, c0, k0, 72, lane);
      acc = wmma32(a, b, acc);
    }
    int col = c0 + n;
    float bias = bo[col];
#pragma unroll
    for (int r = 0; r < 8; ++r)
      xw[(r0 + mb + r) * CC + col] += acc[r] + bias;   // xw becomes x2
  }
  __syncthreads();

  // ---- Phase F: LN2 -> hw; stage w1^T ----
  {
    int tok = tid >> 2, qtr = tid & 3;
    float vals[16];
    float s = 0.f, ss = 0.f;
    const float4* xp = (const float4*)(xw + tok * CC + qtr * 16);
#pragma unroll
    for (int v4 = 0; v4 < 4; ++v4) {
      float4 f = xp[v4];
      vals[v4 * 4 + 0] = f.x; vals[v4 * 4 + 1] = f.y;
      vals[v4 * 4 + 2] = f.z; vals[v4 * 4 + 3] = f.w;
    }
#pragma unroll
    for (int i = 0; i < 16; ++i) { s += vals[i]; ss += vals[i] * vals[i]; }
    s  += __shfl_xor(s, 1);  s  += __shfl_xor(s, 2);
    ss += __shfl_xor(ss, 1); ss += __shfl_xor(ss, 2);
    float mean = s * (1.f / 64.f);
    float var  = ss * (1.f / 64.f) - mean * mean;
    float rstd = rsqrtf(var + 1e-5f);
    h8 o0, o1;
#pragma unroll
    for (int i = 0; i < 8; ++i) {
      int c = qtr * 16 + i;
      o0[i] = (_Float16)((vals[i] - mean) * rstd * ln2_g[c] + ln2_b[c]);
    }
#pragma unroll
    for (int i = 0; i < 8; ++i) {
      int c = qtr * 16 + 8 + i;
      o1[i] = (_Float16)((vals[8 + i] - mean) * rstd * ln2_g[c] + ln2_b[c]);
    }
    *(h8*)(hw + tok * 72 + qtr * 16) = o0;
    *(h8*)(hw + tok * 72 + qtr * 16 + 8) = o1;

    for (int i = tid; i < 64 * 256; i += 256) {
      int r = i >> 8, c = i & 255;                  // r = K (64), c = col (256)
      wt[c * 72 + r] = (_Float16)w1[i];
    }
  }
  __syncthreads();

  // ---- Phase G: MLP1 + exact GELU -> f16 in sb ----
  _Float16* gb = (_Float16*)sb;     // [64][264] halves
  for (int t = wave; t < 64; t += 8) {
    int rt = t >> 4, ct = t & 15;
    int r0 = rt * 16, c0 = ct * 16;
    v8f acc = {};
#pragma unroll
    for (int k0 = 0; k0 < 64; k0 += 32) {
      v16h a = load_a16(hw, r0, k0, 72, lane);
      v16h b = load_bT(wt, c0, k0, 72, lane);
      acc = wmma32(a, b, acc);
    }
    int col = c0 + n;
    float bias = b1[col];
#pragma unroll
    for (int r = 0; r < 8; ++r) {
      float v = acc[r] + bias;
      v = 0.5f * v * (1.f + erff(v * 0.70710678118f));
      gb[(r0 + mb + r) * 264 + col] = (_Float16)v;
    }
  }
  __syncthreads();

  // stage w2^T [col][k], stride 264 (K = 256)
  for (int i = tid; i < 256 * 64; i += 256) {
    int r = i >> 6, c = i & 63;                     // r = K, c = col
    wt[c * 264 + r] = (_Float16)w2[i];
  }
  __syncthreads();

  // ---- Phase H: MLP2 + b2 + residual -> global (window reverse) ----
  for (int t = wave; t < 16; t += 8) {
    int rt = t >> 2, ct = t & 3;
    int r0 = rt * 16, c0 = ct * 16;
    v8f acc = {};
#pragma unroll
    for (int k0 = 0; k0 < 256; k0 += 32) {
      v16h a = load_a16(gb, r0, k0, 264, lane);
      v16h b = load_bT(wt, c0, k0, 264, lane);
      acc = wmma32(a, b, acc);
    }
    int col = c0 + n;
    float bias = b2[col];
#pragma unroll
    for (int r = 0; r < 8; ++r) {
      int row = r0 + mb + r;
      float v = acc[r] + bias + xw[row * CC + col];
      int ty = row >> 3, tx = row & 7;
      size_t g = (((size_t)batch << 16) +
                  (size_t)((wy * 8 + ty) * 256 + (wx * 8 + tx))) * CC + col;
      out[g] = v;
    }
  }
}

extern "C" void kernel_launch(void* const* d_in, const int* in_sizes, int n_in,
                              void* d_out, int out_size, void* d_ws, size_t ws_size,
                              hipStream_t stream) {
  const float* x      = (const float*)d_in[0];
  const float* ln1_g  = (const float*)d_in[1];
  const float* ln1_b  = (const float*)d_in[2];
  const float* wq     = (const float*)d_in[3];
  const float* bq     = (const float*)d_in[4];
  const float* wkv    = (const float*)d_in[5];
  const float* bkv    = (const float*)d_in[6];
  const float* rbt    = (const float*)d_in[7];
  const float* wo     = (const float*)d_in[8];
  const float* bo     = (const float*)d_in[9];
  const float* ln2_g  = (const float*)d_in[10];
  const float* ln2_b  = (const float*)d_in[11];
  const float* w1     = (const float*)d_in[12];
  const float* b1     = (const float*)d_in[13];
  const float* w2     = (const float*)d_in[14];
  const float* b2     = (const float*)d_in[15];
  const int*   relidx = (const int*)d_in[16];
  float* out = (float*)d_out;

  (void)in_sizes; (void)n_in; (void)out_size; (void)d_ws; (void)ws_size;

  hipFuncSetAttribute((const void*)lewin_block_kernel,
                      hipFuncAttributeMaxDynamicSharedMemorySize, SMEM_BYTES);

  lewin_block_kernel<<<NWIN_TOTAL, 256, SMEM_BYTES, stream>>>(
      x, ln1_g, ln1_b, wq, bq, wkv, bkv, rbt, wo, bo,
      ln2_g, ln2_b, w1, b1, w2, b2, relidx, out);
}
